// Gnn_model_68839735821122
// MI455X (gfx1250) — matrix operand order repaired and checked by reading, not compiled, not run on hardware
//
#include <hip/hip_runtime.h>
#include <hip/hip_bf16.h>
#include <math.h>

typedef _Float16 half_t;
typedef __attribute__((ext_vector_type(16))) _Float16 v16h;
typedef __attribute__((ext_vector_type(4)))  _Float16 h4;
typedef __attribute__((ext_vector_type(8)))  float    v8f;

// Padded LDS row lengths: +8 halves (16B) so 16 consecutive rows sweep all
// 64 LDS banks (16 rows x 16B = 256B) -> conflict-free ds_load_b128 fragment
// reads. fp32 rows read column-parallel get +4 floats for the same reason.
constexpr int LD256 = 256 + 8;   // f16 rows with K=256
constexpr int LD128 = 128 + 8;   // f16 rows with K=128
constexpr int LDM   = 128 + 4;   // fp32 rows, 128 cols
constexpr int LDY   = 256 + 4;   // fp32 rows, 256 cols

// LDS budgets (bytes) shared between device layout and host launch
constexpr unsigned EDGE_LDS =
    256 * LD256 * 2 + 128 * LD256 * 2 + (256 + 128 + 128) * 4 +
    32 * LD256 * 2 + 32 * LD256 * 2 + 32 * LDM * 4;                 // 255488
constexpr unsigned NODEPRE_LDS = 384 * LD128 * 2 + 16 * LD128 * 2;  // 108800
constexpr unsigned NODEPOST_LDS =
    384 * LD128 * 2 + 256 * LD128 * 2 + 16 * LD128 * 2 + 16 * LD128 * 2 +
    16 * 384 * 4 + 16 * LDY * 4 + (384 + 256 + 256) * 4;            // 227584

// ---------------------------------------------------------------------------
// WMMA helpers (CDNA5 wave32, 16x16x32 f16 -> f32)
// ---------------------------------------------------------------------------
__device__ __forceinline__ v8f wmma_f16(v16h a, v16h b, v8f c) {
    return __builtin_amdgcn_wmma_f32_16x16x32_f16(false, a, false, b, (short)0, c,
                                                  false, false);
}

// A fragment: 16x32 (MxK) f16, row-major LDS, row stride ld halves.
// Lane L: M=L%16, grp=L/16, K(v,half)=v*2+half+grp*16.
__device__ __forceinline__ v16h load_fragA(const half_t* lds, int row0, int ld, int kc) {
    const int lane = threadIdx.x & 31;
    const int row  = row0 + (lane & 15);
    const int grp  = lane >> 4;
    const half_t* p = lds + row * ld + kc * 32;
    v16h f;
#pragma unroll
    for (int v = 0; v < 8; ++v) {
        const int k = (v << 1) + (grp << 4);
        f[2 * v]     = p[k];
        f[2 * v + 1] = p[k + 1];
    }
    return f;
}

// B fragment: 32x16 (KxN) f16, weights stored K-contiguous per output column.
// Lane L: N=L%16; lanes 0-15 hold K=kc*32+0..15, lanes 16-31 K=kc*32+16..31.
__device__ __forceinline__ v16h load_fragB(const half_t* lds, int col0, int ld, int kc) {
    const int lane = threadIdx.x & 31;
    const int col  = col0 + (lane & 15);
    const int grp  = lane >> 4;
    const half_t* p = lds + col * ld + kc * 32 + grp * 16;
    v16h f;
#pragma unroll
    for (int i = 0; i < 16; ++i) f[i] = p[i];
    return f;
}

// ---------------------------------------------------------------------------
// Kernel 0: fold attention kernels through attn_w:  c = k1·aw1, d2 = k2·aw2
// ---------------------------------------------------------------------------
__global__ void setup_cd_kernel(const float* __restrict__ k1, const float* __restrict__ k2,
                                const float* __restrict__ attn,
                                float* __restrict__ cvec, float* __restrict__ d2v) {
    const int t = threadIdx.x;
    if (t < 128) {
        float a = 0.f, b = 0.f;
        for (int n = 0; n < 128; ++n) {
            a += k1[t * 128 + n] * attn[n];
            b += k2[t * 128 + n] * attn[128 + n];
        }
        cvec[t] = a;
        d2v[t]  = b;
    }
}

// ---------------------------------------------------------------------------
// Kernel 1: row_ptr[n] = lower_bound(dst, n)  (dst is sorted)
// ---------------------------------------------------------------------------
__global__ void rowptr_kernel(const int* __restrict__ dst, int E,
                              int* __restrict__ rowptr, int N) {
    const int n = blockIdx.x * blockDim.x + threadIdx.x;
    if (n > N) return;
    int lo = 0, hi = E;
    while (lo < hi) {
        const int mid = (lo + hi) >> 1;
        if (dst[mid] < n) lo = mid + 1; else hi = mid;
    }
    rowptr[n] = lo;
}

// ---------------------------------------------------------------------------
// Kernel 2: gh = h @ Wh [N,384] (WMMA), tddot[n] = h[n]·d2.  256 thr / 8 waves
// ---------------------------------------------------------------------------
__global__ void __launch_bounds__(256)
nodepre_kernel(const float* __restrict__ h, const float* __restrict__ Wh,
               const float* __restrict__ d2v, float* __restrict__ gh,
               float* __restrict__ tddot, int N, int nTiles) {
    extern __shared__ char smem[];
    half_t* Wht = (half_t*)smem;              // [384][LD128]
    half_t* A   = Wht + 384 * LD128;          // [16][LD128]
    const int t = threadIdx.x, lane = t & 31, wave = t >> 5;

    for (int i = t; i < 384 * 128; i += 256)
        Wht[(i >> 7) * LD128 + (i & 127)] = (half_t)Wh[(size_t)(i & 127) * 384 + (i >> 7)];
    __syncthreads();

    for (int tile = blockIdx.x; tile < nTiles; tile += gridDim.x) {
        const int row0 = tile * 16;
        for (int i = t; i < 16 * 128; i += 256)
            A[(i >> 7) * LD128 + (i & 127)] = (half_t)h[(size_t)(row0 + (i >> 7)) * 128 + (i & 127)];
        __syncthreads();

        v16h af[4];
#pragma unroll
        for (int kc = 0; kc < 4; ++kc) af[kc] = load_fragA(A, 0, LD128, kc);
        for (int nt = wave; nt < 24; nt += 8) {
            v8f acc = {};
#pragma unroll
            for (int kc = 0; kc < 4; ++kc)
                acc = wmma_f16(af[kc], load_fragB(Wht, nt * 16, LD128, kc), acc);
            const int col = nt * 16 + (lane & 15);
            const int g   = lane >> 4;
#pragma unroll
            for (int r = 0; r < 8; ++r)
                gh[(size_t)(row0 + r + 8 * g) * 384 + col] = acc[r];
        }
        if (t < 16) {
            float a = 0.f;
            const float* hr = h + (size_t)(row0 + t) * 128;
            for (int k = 0; k < 128; ++k) a += hr[k] * d2v[k];
            tddot[row0 + t] = a;
        }
        __syncthreads();
    }
}

// ---------------------------------------------------------------------------
// Kernel 3: edge message MLP + attention score. 256 thr, M=32 edges per step:
// two 16-row A-tiles share every B fragment (halves weight LDS traffic).
// ---------------------------------------------------------------------------
__global__ void __launch_bounds__(256)
edge_kernel(const float* __restrict__ h, const int* __restrict__ src,
            const int* __restrict__ dst,
            const float* __restrict__ W1, const float* __restrict__ b1,
            const float* __restrict__ W2, const float* __restrict__ b2,
            const float* __restrict__ cvec, const float* __restrict__ tddot,
            half_t* __restrict__ m16, float* __restrict__ score,
            int E, int nTiles) {
    extern __shared__ char smem[];
    half_t* W1t = (half_t*)smem;               // [256][LD256]
    half_t* W2t = W1t + 256 * LD256;           // [128][LD256]
    float*  b1L = (float*)(W2t + 128 * LD256); // 256
    float*  b2L = b1L + 256;                   // 128
    float*  cL  = b2L + 128;                   // 128
    half_t* A   = (half_t*)(cL + 128);         // [32][LD256]
    half_t* X1  = A + 32 * LD256;              // [32][LD256]
    float*  Mf  = (float*)(X1 + 32 * LD256);   // [32][LDM]
    const int t = threadIdx.x, lane = t & 31, wave = t >> 5;

    for (int i = t; i < 65536; i += 256)
        W1t[(i >> 8) * LD256 + (i & 255)] = (half_t)W1[(size_t)(i & 255) * 256 + (i >> 8)];
    for (int i = t; i < 32768; i += 256)
        W2t[(i >> 8) * LD256 + (i & 255)] = (half_t)W2[(size_t)(i & 255) * 128 + (i >> 8)];
    b1L[t] = b1[t];
    if (t < 128) { b2L[t] = b2[t]; cL[t] = cvec[t]; }
    __syncthreads();

    for (int tile = blockIdx.x; tile < nTiles; tile += gridDim.x) {
        const int e0 = tile * 32;
        // gather concat(h_src, h_dst) as f16 (coalesced 512B row segments)
        for (int i = t; i < 32 * 256; i += 256) {
            const int row = i >> 8, col = i & 255;
            const int e = e0 + row;
            const int node = (col < 128) ? src[e] : dst[e];
            A[row * LD256 + col] = (half_t)h[(size_t)node * 128 + (col & 127)];
        }
        __syncthreads();

        // layer 1: X1 = relu(A @ W1 + b1)  16 N-tiles, K=256, 2 M-tiles
        {
            v16h a0[8], a1[8];
#pragma unroll
            for (int kc = 0; kc < 8; ++kc) {
                a0[kc] = load_fragA(A,  0, LD256, kc);
                a1[kc] = load_fragA(A, 16, LD256, kc);
            }
            for (int nt = wave; nt < 16; nt += 8) {
                v8f acc0 = {}, acc1 = {};
#pragma unroll
                for (int kc = 0; kc < 8; ++kc) {
                    const v16h b = load_fragB(W1t, nt * 16, LD256, kc);
                    acc0 = wmma_f16(a0[kc], b, acc0);
                    acc1 = wmma_f16(a1[kc], b, acc1);
                }
                const int col = nt * 16 + (lane & 15);
                const int g   = lane >> 4;
#pragma unroll
                for (int r = 0; r < 8; ++r) {
                    X1[(r + 8 * g) * LD256 + col]        = (half_t)fmaxf(acc0[r] + b1L[col], 0.f);
                    X1[(16 + r + 8 * g) * LD256 + col]   = (half_t)fmaxf(acc1[r] + b1L[col], 0.f);
                }
            }
        }
        __syncthreads();

        // layer 2: Mf = X1 @ W2 + b2  8 N-tiles, K=256, 2 M-tiles
        {
            v16h a0[8], a1[8];
#pragma unroll
            for (int kc = 0; kc < 8; ++kc) {
                a0[kc] = load_fragA(X1,  0, LD256, kc);
                a1[kc] = load_fragA(X1, 16, LD256, kc);
            }
            const int nt = wave;   // 8 waves cover 8 N-tiles
            v8f acc0 = {}, acc1 = {};
#pragma unroll
            for (int kc = 0; kc < 8; ++kc) {
                const v16h b = load_fragB(W2t, nt * 16, LD256, kc);
                acc0 = wmma_f16(a0[kc], b, acc0);
                acc1 = wmma_f16(a1[kc], b, acc1);
            }
            const int col = nt * 16 + (lane & 15);
            const int g   = lane >> 4;
#pragma unroll
            for (int r = 0; r < 8; ++r) {
                Mf[(r + 8 * g) * LDM + col]      = acc0[r] + b2L[col];
                Mf[(16 + r + 8 * g) * LDM + col] = acc1[r] + b2L[col];
            }
        }
        __syncthreads();

        // emit m (f16, 8B vector stores) and attention scores
        for (int i = t; i < 32 * 32; i += 256) {
            const int row = i >> 5, c4 = (i & 31) << 2;
            h4 v;
#pragma unroll
            for (int j = 0; j < 4; ++j) v[j] = (half_t)Mf[row * LDM + c4 + j];
            *(h4*)(m16 + (size_t)(e0 + row) * 128 + c4) = v;
        }
        if (t < 32) {
            float s = 0.f;
            for (int k = 0; k < 128; ++k) s += Mf[t * LDM + k] * cL[k];
            s += tddot[dst[e0 + t]];
            score[e0 + t] = (s > 0.f) ? s : 0.2f * s;   // leaky_relu(0.2)
        }
        __syncthreads();
    }
}

// ---------------------------------------------------------------------------
// Kernel 4: per-destination softmax + weighted sum (segments contiguous).
// ---------------------------------------------------------------------------
__global__ void __launch_bounds__(128)
aggregate_kernel(const float* __restrict__ score, const half_t* __restrict__ m16,
                 const int* __restrict__ rowptr, float* __restrict__ agg) {
    __shared__ float red[128];
    __shared__ float alpha[128];
    const int n = blockIdx.x;
    const int t = threadIdx.x;
    const int s0 = rowptr[n], s1 = rowptr[n + 1];

    float local = -INFINITY;
    for (int e = s0 + t; e < s1; e += 128) local = fmaxf(local, score[e]);
    red[t] = local; __syncthreads();
    for (int s = 64; s > 0; s >>= 1) { if (t < s) red[t] = fmaxf(red[t], red[t + s]); __syncthreads(); }
    const float smax = red[0]; __syncthreads();

    local = 0.f;
    for (int e = s0 + t; e < s1; e += 128) local += expf(score[e] - smax);
    red[t] = local; __syncthreads();
    for (int s = 64; s > 0; s >>= 1) { if (t < s) red[t] += red[t + s]; __syncthreads(); }
    const float inv = 1.f / (red[0] + 1e-9f); __syncthreads();

    float acc = 0.f;
    for (int chunk = s0; chunk < s1; chunk += 128) {
        const int cnt = min(128, s1 - chunk);
        if (t < cnt) alpha[t] = expf(score[chunk + t] - smax) * inv;
        __syncthreads();
        for (int j = 0; j < cnt; ++j)
            acc += alpha[j] * (float)m16[(size_t)(chunk + j) * 128 + t];
        __syncthreads();
    }
    agg[(size_t)n * 128 + t] = acc;
}

// ---------------------------------------------------------------------------
// Kernel 5: gx = agg@Wx + bg ; GRU(h, gh, gx) -> h_new ; readout MLP -> out
// ---------------------------------------------------------------------------
__global__ void __launch_bounds__(256)
nodepost_kernel(const float* __restrict__ agg, const float* __restrict__ h,
                const float* __restrict__ gh,
                const float* __restrict__ Wx, const float* __restrict__ bg,
                const float* __restrict__ R1, const float* __restrict__ rb1,
                const float* __restrict__ R2, const float* __restrict__ rb2,
                float* __restrict__ out, int N, int nTiles) {
    extern __shared__ char smem[];
    half_t* Wxt  = (half_t*)smem;              // [384][LD128]
    half_t* R1t  = Wxt + 384 * LD128;          // [256][LD128]
    half_t* Aagg = R1t + 256 * LD128;          // [16][LD128]
    half_t* Hn   = Aagg + 16 * LD128;          // [16][LD128]
    float*  GX   = (float*)(Hn + 16 * LD128);  // [16][384]
    float*  Y    = GX + 16 * 384;              // [16][LDY]
    float*  bgL  = Y + 16 * LDY;               // 384
    float*  rb1L = bgL + 384;                  // 256
    float*  R2L  = rb1L + 256;                 // 256
    const int t = threadIdx.x, lane = t & 31, wave = t >> 5;

    for (int i = t; i < 49152; i += 256)
        Wxt[(i >> 7) * LD128 + (i & 127)] = (half_t)Wx[(size_t)(i & 127) * 384 + (i >> 7)];
    for (int i = t; i < 32768; i += 256)
        R1t[(i >> 7) * LD128 + (i & 127)] = (half_t)R1[(size_t)(i & 127) * 256 + (i >> 7)];
    for (int i = t; i < 384; i += 256) bgL[i] = bg[i];
    if (t < 256) { rb1L[t] = rb1[t]; R2L[t] = R2[t]; }
    __syncthreads();

    for (int tile = blockIdx.x; tile < nTiles; tile += gridDim.x) {
        const int row0 = tile * 16;
        for (int i = t; i < 16 * 128; i += 256)
            Aagg[(i >> 7) * LD128 + (i & 127)] =
                (half_t)agg[(size_t)(row0 + (i >> 7)) * 128 + (i & 127)];
        __syncthreads();

        // gx = agg @ Wx + bg  (24 N-tiles, K=128)
        {
            v16h af[4];
#pragma unroll
            for (int kc = 0; kc < 4; ++kc) af[kc] = load_fragA(Aagg, 0, LD128, kc);
            for (int nt = wave; nt < 24; nt += 8) {
                v8f acc = {};
#pragma unroll
                for (int kc = 0; kc < 4; ++kc)
                    acc = wmma_f16(af[kc], load_fragB(Wxt, nt * 16, LD128, kc), acc);
                const int col = nt * 16 + (lane & 15);
                const int g   = lane >> 4;
#pragma unroll
                for (int r = 0; r < 8; ++r)
                    GX[(r + 8 * g) * 384 + col] = acc[r] + bgL[col];
            }
        }
        __syncthreads();

        // GRU elementwise -> Hn (f16)
        for (int i = t; i < 16 * 128; i += 256) {
            const int row = i >> 7, f = i & 127;
            const size_t node = (size_t)(row0 + row);
            const float xz = GX[row * 384 + f];
            const float xr = GX[row * 384 + 128 + f];
            const float xn = GX[row * 384 + 256 + f];
            const float hz = gh[node * 384 + f];
            const float hr = gh[node * 384 + 128 + f];
            const float hn = gh[node * 384 + 256 + f];
            const float z  = 1.f / (1.f + expf(-(xz + hz)));
            const float rr = 1.f / (1.f + expf(-(xr + hr)));
            const float cand = tanhf(xn + rr * hn);
            const float hv = h[node * 128 + f];
            Hn[row * LD128 + f] = (half_t)(z * hv + (1.f - z) * cand);
        }
        __syncthreads();

        // Y = relu(h_new @ R1 + rb1)  (16 N-tiles, K=128)
        {
            v16h af[4];
#pragma unroll
            for (int kc = 0; kc < 4; ++kc) af[kc] = load_fragA(Hn, 0, LD128, kc);
            for (int nt = wave; nt < 16; nt += 8) {
                v8f acc = {};
#pragma unroll
                for (int kc = 0; kc < 4; ++kc)
                    acc = wmma_f16(af[kc], load_fragB(R1t, nt * 16, LD128, kc), acc);
                const int col = nt * 16 + (lane & 15);
                const int g   = lane >> 4;
#pragma unroll
                for (int r = 0; r < 8; ++r)
                    Y[(r + 8 * g) * LDY + col] = fmaxf(acc[r] + rb1L[col], 0.f);
            }
        }
        __syncthreads();

        if (t < 16) {
            float a = rb2[0];
            for (int j = 0; j < 256; ++j) a += Y[t * LDY + j] * R2L[j];
            out[row0 + t] = a;
        }
        __syncthreads();
    }
}

// ---------------------------------------------------------------------------
// Host launcher
// ---------------------------------------------------------------------------
extern "C" void kernel_launch(void* const* d_in, const int* in_sizes, int n_in,
                              void* d_out, int out_size, void* d_ws, size_t ws_size,
                              hipStream_t stream) {
    const float* h    = (const float*)d_in[0];
    const int*   src  = (const int*)d_in[1];
    const int*   dst  = (const int*)d_in[2];
    const float* W1   = (const float*)d_in[3];
    const float* b1   = (const float*)d_in[4];
    const float* W2   = (const float*)d_in[5];
    const float* b2   = (const float*)d_in[6];
    const float* k1   = (const float*)d_in[7];
    const float* k2   = (const float*)d_in[8];
    const float* attn = (const float*)d_in[9];
    const float* Wx   = (const float*)d_in[10];
    const float* Wh   = (const float*)d_in[11];
    const float* bg   = (const float*)d_in[12];
    const float* R1   = (const float*)d_in[13];
    const float* rb1  = (const float*)d_in[14];
    const float* R2   = (const float*)d_in[15];
    const float* rb2  = (const float*)d_in[16];
    float* out = (float*)d_out;
    (void)n_in; (void)out_size; (void)ws_size;

    const int N = in_sizes[0] / 128;   // 50000 (divisible by 16)
    const int E = in_sizes[1];         // 800000 (divisible by 32)
    const int nodeTiles  = (N + 15) / 16;
    const int edgeTiles  = (E + 31) / 32;

    // workspace carve (256B aligned)
    char* w = (char*)d_ws;
    auto carve = [&](size_t bytes) { void* p = (void*)w; w += (bytes + 255) & ~(size_t)255; return p; };
    float*  cvec   = (float*)carve(128 * sizeof(float));
    float*  d2v    = (float*)carve(128 * sizeof(float));
    float*  tddot  = (float*)carve((size_t)N * sizeof(float));
    int*    rowptr = (int*)carve((size_t)(N + 1) * sizeof(int));
    float*  gh     = (float*)carve((size_t)N * 384 * sizeof(float));
    float*  score  = (float*)carve((size_t)E * sizeof(float));
    float*  agg    = (float*)carve((size_t)N * 128 * sizeof(float));
    half_t* m16    = (half_t*)carve((size_t)E * 128 * sizeof(half_t));

    hipLaunchKernelGGL(setup_cd_kernel, dim3(1), dim3(128), 0, stream,
                       k1, k2, attn, cvec, d2v);
    hipLaunchKernelGGL(rowptr_kernel, dim3((N + 1 + 255) / 256), dim3(256), 0, stream,
                       dst, E, rowptr, N);
    hipLaunchKernelGGL(nodepre_kernel, dim3(512), dim3(256), NODEPRE_LDS, stream,
                       h, Wh, d2v, gh, tddot, N, nodeTiles);
    hipLaunchKernelGGL(edge_kernel, dim3(2048), dim3(256), EDGE_LDS, stream,
                       h, src, dst, W1, b1, W2, b2, cvec, tddot, m16, score, E, edgeTiles);
    hipLaunchKernelGGL(aggregate_kernel, dim3(N), dim3(128), 0, stream,
                       score, m16, rowptr, agg);
    hipLaunchKernelGGL(nodepost_kernel, dim3(512), dim3(256), NODEPOST_LDS, stream,
                       agg, h, gh, Wx, bg, R1, rb1, R2, rb2, out, N, nodeTiles);
}
